// PFLM_45208825757911
// MI455X (gfx1250) — matrix operand order repaired and checked by reading, not compiled, not run on hardware
//
#include <hip/hip_runtime.h>
#include <hip/hip_bf16.h>
#include <stdint.h>

// =====================================================================
// Seq-VAE LM forward on gfx1250 (MI455X), bf16 WMMA everywhere.
//   T=64, B=32 -> M=2048 rows. NINP=256, NHID=512, ZDIM=128, V=32000.
// Round 3 changes vs round 2:
//   * GEMM W staging now uses gfx1250 async data mover:
//     global_load_async_to_lds_b128 (+ s_wait_asynccnt) instead of the
//     global_load/ds_store round trip through VGPRs.
//   * __builtin_amdgcn_sched_barrier(0) between fragment-load groups and
//     WMMA groups: forces all B fragments into distinct live VGPRs so
//     the independent WMMAs issue back-to-back behind a single wait
//     (round-2 asm showed the allocator coalescing b[] into one pair,
//     serializing ds_load/wait/wmma).
// =====================================================================

typedef __attribute__((ext_vector_type(16))) __bf16 bf16x16;
typedef __attribute__((ext_vector_type(8)))  float  f32x8;

union FragU { uint4 q[2]; bf16x16 f; };

__device__ __forceinline__ unsigned short f2bf(float x) {
  unsigned u = __builtin_bit_cast(unsigned, x);
  u = (u + 0x7FFFu + ((u >> 16) & 1u)) >> 16;   // RNE
  return (unsigned short)u;
}

// A-matrix 16x32 bf16 fragment (row-major source; p = row ptr at k-chunk
// start, u32 units). Lane-half h: regs 0..3 hold K = 8h+{0..7},
// regs 4..7 hold K = 16+8h+{0..7} -> u32 {4h..4h+3} and {8+4h..11+4h}.
__device__ __forceinline__ bf16x16 load_frag_a(const unsigned* p, int kh) {
  FragU u;
  u.q[0] = *(const uint4*)(p + 4 * kh);
  u.q[1] = *(const uint4*)(p + 8 + 4 * kh);
  return u.f;
}

// B-matrix 32x16 bf16 fragment from W[N][K] row-major (B = W^T). Lane's
// row ptr at k-chunk start; same K order as A -> u32 {4h..4h+3}, {8+4h..11+4h}.
__device__ __forceinline__ bf16x16 load_frag_b(const unsigned* p, int kh) {
  FragU u;
  u.q[0] = *(const uint4*)(p + 4 * kh);
  u.q[1] = *(const uint4*)(p + 8 + 4 * kh);
  return u.f;
}

__device__ __forceinline__ f32x8 wmma_bf16(bf16x16 a, bf16x16 b, f32x8 c) {
  return __builtin_amdgcn_wmma_f32_16x16x32_bf16(false, a, false, b,
                                                 (short)0, c, false, false);
}

__device__ __forceinline__ float sigf(float x) {
  return 1.0f / (1.0f + __expf(-x));
}

// gfx1250 async data mover: copy 16B/lane global -> LDS, ASYNCcnt-tracked.
// lds_byte = LDS byte offset (low 32 bits of the generic pointer: the
// aperture mapping truncates flat shared addresses to addr[31:0]).
__device__ __forceinline__ void async_b128(unsigned lds_byte,
                                           const void* gsrc) {
  asm volatile("global_load_async_to_lds_b128 %0, %1, off"
               :: "v"(lds_byte), "v"(gsrc) : "memory");
}
__device__ __forceinline__ void wait_async0() {
  asm volatile("s_wait_asynccnt 0x0" ::: "memory");
}

// ---------------------------------------------------------------------
// fp32 -> bf16 conversion
// ---------------------------------------------------------------------
__global__ void cvt_kernel(const float* __restrict__ in,
                           unsigned short* __restrict__ out, int n) {
  int i = blockIdx.x * blockDim.x + threadIdx.x;
  if (i < n) out[i] = f2bf(in[i]);
}

// ---------------------------------------------------------------------
// Embedding gathers: encoder emb (stride 256) and decoder emb into the
// concat buffer [.,384] cols 0..255 (z fills 256..383 later).
// ---------------------------------------------------------------------
__global__ void embed_kernel(const int* __restrict__ toks,
                             const int* __restrict__ tgts,
                             const float* __restrict__ iemb,
                             const float* __restrict__ demb,
                             unsigned short* __restrict__ emb_bf,
                             unsigned short* __restrict__ dec_in) {
  const int row = blockIdx.x;      // t*32 + b, 0..2047
  const int c = threadIdx.x;       // 0..255
  const int t1 = toks[row];
  const int t2 = tgts[row];
  emb_bf[(size_t)row * 256 + c] = f2bf(iemb[(size_t)t1 * 256 + c]);
  dec_in[(size_t)row * 384 + c] = f2bf(demb[(size_t)t2 * 256 + c]);
}

// ---------------------------------------------------------------------
// Generic WMMA GEMM: C[M,N] = A[M,K](bf16) @ W[N,K]^T(bf16) + bias[N].
// 256 threads = 8 waves; block tile 128(M) x 128(N); wave = one 16-row
// M stripe x eight 16-col N tiles. W k-slab (128x32 bf16 = 8KB) staged
// into LDS via ASYNC loads, double-buffered: stage(k+1) overlaps
// compute(k); drain with s_wait_asynccnt before the single barrier.
// Grid: (N/128, M/128). K multiple of 32.
// ---------------------------------------------------------------------
__global__ __launch_bounds__(256) void gemm_bf16_kernel(
    const unsigned short* __restrict__ A, const unsigned short* __restrict__ W,
    const float* __restrict__ bias, float* __restrict__ C, int N, int K) {
  extern __shared__ unsigned smw[];  // 2 buffers x 2048 u32 (8KB each)
  const int tid = threadIdx.x;
  const int wave = tid >> 5, lane = tid & 31;
  const int lm = lane & 15, lh = lane >> 4;
  const int nb = blockIdx.x, mb = blockIdx.y;
  const int Ku = K >> 1;  // u32 per row

  f32x8 acc[8];
  for (int nt = 0; nt < 8; ++nt) {
    const float bv = bias[nb * 128 + nt * 16 + lm];
    for (int r = 0; r < 8; ++r) acc[nt][r] = bv;
  }

  const unsigned* Arow = (const unsigned*)(const void*)A +
                         (size_t)(mb * 128 + wave * 16 + lm) * Ku;
  const unsigned* Wp = (const unsigned*)(const void*)W;
  // Each thread stages 8 consecutive u32 (half a 32-elem row) per chunk.
  const int srow = (tid * 8) >> 4;   // 0..127
  const int scu = (tid * 8) & 15;    // 0 or 8
  const unsigned* Wstage = Wp + (size_t)(nb * 128 + srow) * Ku + scu;
  const unsigned smw_lds = (unsigned)(uintptr_t)(void*)smw;  // LDS byte base

  auto stage = [&](int kc, int buf) {
    const unsigned dst = smw_lds + (unsigned)(buf * 8192 + tid * 32);
    async_b128(dst, (const void*)(Wstage + kc * 16));
    async_b128(dst + 16, (const void*)(Wstage + kc * 16 + 4));
  };

  const int KC = K >> 5;
  stage(0, 0);
  wait_async0();
  __syncthreads();
  for (int kc = 0; kc < KC; ++kc) {
    const int buf = kc & 1;
    if (kc + 1 < KC) stage(kc + 1, buf ^ 1);   // async, overlaps compute
    bf16x16 a = load_frag_a(Arow + kc * 16, lh);
    const unsigned* bbase = smw + buf * 2048 + lm * 16;
    bf16x16 b[8];
#pragma unroll
    for (int nt = 0; nt < 8; ++nt)
      b[nt] = load_frag_b(bbase + nt * 256, lh);
    __builtin_amdgcn_sched_barrier(0);  // keep all 8 B frags live
#pragma unroll
    for (int nt = 0; nt < 8; ++nt)
      acc[nt] = wmma_bf16(a, b[nt], acc[nt]);
    wait_async0();     // next-buffer staging complete (own wave's ops)
    __syncthreads();   // visible to all waves; reads of buf done
  }

  for (int nt = 0; nt < 8; ++nt) {
    const int col = nb * 128 + nt * 16 + lm;
    for (int r = 0; r < 8; ++r) {
      const int row = mb * 128 + wave * 16 + r + 8 * lh;
      C[(size_t)row * N + col] = acc[nt][r];
    }
  }
}

// ---------------------------------------------------------------------
// LSTM scan. grid = 1 or 2 blocks; each block runs an independent scan
// (per-block params -> fwd and bwd encoder directions run concurrently).
// 1024 thr = 32 waves. LDS: c fp32[32][512] (64KB) + h bf16[32][512]
// (32KB). gates[32,2048] = xg[t] + h @ Whh^T via WMMA; each wave owns
// matching i/f/g/o columns {g*512+j}, so the nonlinearity runs entirely
// from accumulator registers — no gate staging, 2 barriers per step.
// ---------------------------------------------------------------------
__global__ __launch_bounds__(1024) void lstm_scan_kernel(
    const float* __restrict__ xg0, const unsigned short* __restrict__ Whh0,
    unsigned short* hout0, int stride0, int coloff0, int rev0,
    const float* __restrict__ xg1, const unsigned short* __restrict__ Whh1,
    unsigned short* hout1, int stride1, int coloff1, int rev1, int T) {
  const float* xg;
  const unsigned short* Whh;
  unsigned short* hout;
  int out_stride, out_coloff, reverse;
  if (blockIdx.x == 0) {
    xg = xg0; Whh = Whh0; hout = hout0;
    out_stride = stride0; out_coloff = coloff0; reverse = rev0;
  } else {
    xg = xg1; Whh = Whh1; hout = hout1;
    out_stride = stride1; out_coloff = coloff1; reverse = rev1;
  }

  extern __shared__ char smraw[];
  float* cst = (float*)smraw;                                    // [32][512]
  unsigned short* hb = (unsigned short*)(smraw + 32 * 512 * 4);  // [32][512]
  const int tid = threadIdx.x;
  for (int i = tid; i < 32 * 512; i += 1024) { cst[i] = 0.0f; hb[i] = 0; }

  const int wave = tid >> 5, lane = tid & 31;
  const int lm = lane & 15, lh = lane >> 4;
  const int mtile = wave & 1;          // 2 M tiles cover B=32
  const int jbase = (wave >> 1) * 32;  // 16 groups x 32 hidden cols = 512
  const unsigned* Wp = (const unsigned*)(const void*)Whh;

  for (int step = 0; step < T; ++step) {
    const int t = reverse ? (T - 1 - step) : step;
    const float* xgt = xg + (size_t)t * 32 * 2048;
    __syncthreads();  // h from previous step fully written

    f32x8 acc[8];  // [jt][gate]
#pragma unroll
    for (int jt = 0; jt < 2; ++jt)
#pragma unroll
      for (int g = 0; g < 4; ++g) {
        const int col = g * 512 + jbase + jt * 16 + lm;
        for (int r = 0; r < 8; ++r)
          acc[jt * 4 + g][r] = xgt[(mtile * 16 + r + 8 * lh) * 2048 + col];
      }

    const unsigned* hrow =
        (const unsigned*)(const void*)hb + (mtile * 16 + lm) * 256;
    for (int kc = 0; kc < 16; ++kc) {  // K = 512
      bf16x16 a = load_frag_a(hrow + kc * 16, lh);
#pragma unroll
      for (int jt = 0; jt < 2; ++jt) {
        bf16x16 b[4];  // preload 4 B frags -> 4 back-to-back WMMAs
#pragma unroll
        for (int g = 0; g < 4; ++g)
          b[g] = load_frag_b(
              Wp + (size_t)(g * 512 + jbase + jt * 16 + lm) * 256 + kc * 16,
              lh);
        __builtin_amdgcn_sched_barrier(0);  // keep 4 B frags live
#pragma unroll
        for (int g = 0; g < 4; ++g)
          acc[jt * 4 + g] = wmma_bf16(a, b[g], acc[jt * 4 + g]);
      }
    }
    __syncthreads();  // all reads of h done before overwrite

#pragma unroll
    for (int jt = 0; jt < 2; ++jt) {
      const int j = jbase + jt * 16 + lm;
      for (int r = 0; r < 8; ++r) {
        const int m = mtile * 16 + r + 8 * lh;
        const float ig = sigf(acc[jt * 4 + 0][r]);
        const float fg = sigf(acc[jt * 4 + 1][r]);
        const float gg = tanhf(acc[jt * 4 + 2][r]);
        const float og = sigf(acc[jt * 4 + 3][r]);
        const float cn = fg * cst[m * 512 + j] + ig * gg;
        const float h = og * tanhf(cn);
        cst[m * 512 + j] = cn;  // lane-exclusive cell
        const unsigned short hv = f2bf(h);
        hb[m * 512 + j] = hv;
        hout[(size_t)(t * 32 + m) * out_stride + out_coloff + j] = hv;
      }
    }
  }
}

// ---------------------------------------------------------------------
// z scan: LSTMCell(1024->128) + mean/logvar proj + reparameterization.
// 512 threads = 16 waves: mtile = w&1, j-tile = w>>1 (8 x 16 = 128 cols).
// LDS: c fp32[32][128] + z(bf16)[32][128] + hl(bf16)[32][128] = 32KB.
// ---------------------------------------------------------------------
__global__ __launch_bounds__(512) void z_scan_kernel(
    const float* __restrict__ zxg,            // [T][32][512]
    const unsigned short* __restrict__ Whh,   // [512][128] bf16
    const unsigned short* __restrict__ mW,    // [128][128] bf16
    const unsigned short* __restrict__ vW,    // [128][128] bf16
    const float* __restrict__ mb, const float* __restrict__ vb,
    const float* __restrict__ eps,            // [T*32][128]
    unsigned short* __restrict__ dec_in,      // [T*32][384], z at col 256
    float* __restrict__ mean_out, float* __restrict__ lv_out, int T) {
  extern __shared__ char smraw[];
  float* cst = (float*)smraw;                                    // [32][128]
  unsigned short* hz = (unsigned short*)(smraw + 32 * 128 * 4);  // recurrent z
  unsigned short* hl = hz + 32 * 128;                            // cell output
  const int tid = threadIdx.x;
  for (int i = tid; i < 32 * 128; i += 512) {
    cst[i] = 0.0f; hz[i] = 0; hl[i] = 0;
  }

  const int wave = tid >> 5, lane = tid & 31;
  const int lm = lane & 15, lh = lane >> 4;
  const int mtile = wave & 1;
  const int col = (wave >> 1) * 16 + lm;  // 0..127
  const unsigned* Wp = (const unsigned*)(const void*)Whh;
  const unsigned* mWp = (const unsigned*)(const void*)mW;
  const unsigned* vWp = (const unsigned*)(const void*)vW;

  for (int t = 0; t < T; ++t) {
    __syncthreads();  // z from previous step visible
    const float* xgt = zxg + (size_t)t * 32 * 512;
    f32x8 acc[4];
#pragma unroll
    for (int g = 0; g < 4; ++g)
      for (int r = 0; r < 8; ++r)
        acc[g][r] = xgt[(mtile * 16 + r + 8 * lh) * 512 + g * 128 + col];

    const unsigned* hrow =
        (const unsigned*)(const void*)hz + (mtile * 16 + lm) * 64;
#pragma unroll
    for (int kc = 0; kc < 4; ++kc) {  // K = 128
      bf16x16 a = load_frag_a(hrow + kc * 16, lh);
      bf16x16 b[4];
#pragma unroll
      for (int g = 0; g < 4; ++g)
        b[g] = load_frag_b(Wp + (size_t)(g * 128 + col) * 64 + kc * 16, lh);
      __builtin_amdgcn_sched_barrier(0);
#pragma unroll
      for (int g = 0; g < 4; ++g) acc[g] = wmma_bf16(a, b[g], acc[g]);
    }
    for (int r = 0; r < 8; ++r) {  // lane-exclusive cells -> hl
      const int m = mtile * 16 + r + 8 * lh;
      const float ig = sigf(acc[0][r]);
      const float fg = sigf(acc[1][r]);
      const float gg = tanhf(acc[2][r]);
      const float og = sigf(acc[3][r]);
      const float cn = fg * cst[m * 128 + col] + ig * gg;
      cst[m * 128 + col] = cn;
      hl[m * 128 + col] = f2bf(og * tanhf(cn));
    }
    __syncthreads();  // hl ready; all reads of hz done

    f32x8 am, av;
    for (int r = 0; r < 8; ++r) { am[r] = mb[col]; av[r] = vb[col]; }
    const unsigned* hlrow =
        (const unsigned*)(const void*)hl + (mtile * 16 + lm) * 64;
#pragma unroll
    for (int kc = 0; kc < 4; ++kc) {
      bf16x16 a = load_frag_a(hlrow + kc * 16, lh);
      bf16x16 b1 = load_frag_b(mWp + (size_t)col * 64 + kc * 16, lh);
      bf16x16 b2 = load_frag_b(vWp + (size_t)col * 64 + kc * 16, lh);
      __builtin_amdgcn_sched_barrier(0);
      am = wmma_bf16(a, b1, am);
      av = wmma_bf16(a, b2, av);
    }
    for (int r = 0; r < 8; ++r) {
      const int m = mtile * 16 + r + 8 * lh;
      const size_t row = (size_t)t * 32 + m;
      const float mn = am[r], lv = av[r];
      const float z = eps[row * 128 + col] * __expf(0.5f * lv) + mn;
      mean_out[row * 128 + col] = mn;
      lv_out[row * 128 + col] = lv;
      hz[m * 128 + col] = f2bf(z);              // recurrent h for t+1
      dec_in[row * 384 + 256 + col] = f2bf(z);  // decoder concat input
    }
  }
}

// ---------------------------------------------------------------------
// Host-side orchestration
// ---------------------------------------------------------------------
extern "C" void kernel_launch(void* const* d_in, const int* in_sizes, int n_in,
                              void* d_out, int out_size, void* d_ws,
                              size_t ws_size, hipStream_t stream) {
  (void)in_sizes; (void)n_in; (void)out_size; (void)ws_size;
  const int* toks        = (const int*)d_in[0];
  const int* tgts        = (const int*)d_in[1];
  const float* eps       = (const float*)d_in[2];
  const float* inp_emb   = (const float*)d_in[3];
  const float* dec_emb   = (const float*)d_in[4];
  const float* enc_Wih_f = (const float*)d_in[5];
  const float* enc_Whh_f = (const float*)d_in[6];
  const float* enc_b_f   = (const float*)d_in[7];
  const float* enc_Wih_b = (const float*)d_in[8];
  const float* enc_Whh_b = (const float*)d_in[9];
  const float* enc_b_b   = (const float*)d_in[10];
  const float* z_Wih     = (const float*)d_in[11];
  const float* z_Whh     = (const float*)d_in[12];
  const float* z_b       = (const float*)d_in[13];
  const float* mean_W    = (const float*)d_in[14];
  const float* mean_b    = (const float*)d_in[15];
  const float* logvar_W  = (const float*)d_in[16];
  const float* logvar_b  = (const float*)d_in[17];
  const float* dec_Wih   = (const float*)d_in[18];
  const float* dec_Whh   = (const float*)d_in[19];
  const float* dec_b     = (const float*)d_in[20];
  const float* out_W     = (const float*)d_in[21];
  const float* out_b     = (const float*)d_in[22];

  float* logits = (float*)d_out;                  // [2048][32000]
  float* means  = logits + (size_t)2048 * 32000;  // [2048][128]
  float* lvs    = means + (size_t)2048 * 128;     // [2048][128]

  char* ws = (char*)d_ws;
  size_t off = 0;
  auto alloc = [&](size_t bytes) -> void* {
    void* p = ws + off;
    off += (bytes + 255) & ~(size_t)255;
    return p;
  };
  float* xg_f   = (float*)alloc((size_t)2048 * 2048 * 4);
  float* xg_b   = (float*)alloc((size_t)2048 * 2048 * 4);
  float* dec_xg = (float*)alloc((size_t)2048 * 2048 * 4);
  float* zxg    = (float*)alloc((size_t)2048 * 512 * 4);
  unsigned short* emb_bf    = (unsigned short*)alloc((size_t)2048 * 256 * 2);
  unsigned short* hs_bf     = (unsigned short*)alloc((size_t)2048 * 1024 * 2);
  unsigned short* dec_in_bf = (unsigned short*)alloc((size_t)2048 * 384 * 2);
  unsigned short* dec_h_bf  = (unsigned short*)alloc((size_t)2048 * 512 * 2);
  unsigned short* wihf_bf = (unsigned short*)alloc((size_t)2048 * 256 * 2);
  unsigned short* whhf_bf = (unsigned short*)alloc((size_t)2048 * 512 * 2);
  unsigned short* wihb_bf = (unsigned short*)alloc((size_t)2048 * 256 * 2);
  unsigned short* whhb_bf = (unsigned short*)alloc((size_t)2048 * 512 * 2);
  unsigned short* zwih_bf = (unsigned short*)alloc((size_t)512 * 1024 * 2);
  unsigned short* zwhh_bf = (unsigned short*)alloc((size_t)512 * 128 * 2);
  unsigned short* mw_bf   = (unsigned short*)alloc((size_t)128 * 128 * 2);
  unsigned short* vw_bf   = (unsigned short*)alloc((size_t)128 * 128 * 2);
  unsigned short* dwih_bf = (unsigned short*)alloc((size_t)2048 * 384 * 2);
  unsigned short* dwhh_bf = (unsigned short*)alloc((size_t)2048 * 512 * 2);
  unsigned short* outw_bf = (unsigned short*)alloc((size_t)32000 * 512 * 2);

  auto cvt = [&](const float* src, unsigned short* dst, int n) {
    cvt_kernel<<<(n + 255) / 256, 256, 0, stream>>>(src, dst, n);
  };
  cvt(enc_Wih_f, wihf_bf, 2048 * 256);
  cvt(enc_Whh_f, whhf_bf, 2048 * 512);
  cvt(enc_Wih_b, wihb_bf, 2048 * 256);
  cvt(enc_Whh_b, whhb_bf, 2048 * 512);
  cvt(z_Wih, zwih_bf, 512 * 1024);
  cvt(z_Whh, zwhh_bf, 512 * 128);
  cvt(mean_W, mw_bf, 128 * 128);
  cvt(logvar_W, vw_bf, 128 * 128);
  cvt(dec_Wih, dwih_bf, 2048 * 384);
  cvt(dec_Whh, dwhh_bf, 2048 * 512);
  cvt(out_W, outw_bf, 32000 * 512);

  embed_kernel<<<2048, 256, 0, stream>>>(toks, tgts, inp_emb, dec_emb,
                                         emb_bf, dec_in_bf);

  // Encoder input projections: [2048,256] @ [2048,256]^T -> [2048,2048]
  gemm_bf16_kernel<<<dim3(16, 16), 256, 16384, stream>>>(
      emb_bf, wihf_bf, enc_b_f, xg_f, 2048, 256);
  gemm_bf16_kernel<<<dim3(16, 16), 256, 16384, stream>>>(
      emb_bf, wihb_bf, enc_b_b, xg_b, 2048, 256);

  // Encoder scans, BOTH DIRECTIONS IN ONE LAUNCH (grid=2):
  // fwd -> hs_bf cols 0..511, bwd -> cols 512..1023.
  lstm_scan_kernel<<<2, 1024, 96 * 1024, stream>>>(
      xg_f, whhf_bf, hs_bf, 1024, 0, 0,
      xg_b, whhb_bf, hs_bf, 1024, 512, 1, 64);

  // z input projection: [2048,1024] @ [512,1024]^T -> [2048,512]
  gemm_bf16_kernel<<<dim3(4, 16), 256, 16384, stream>>>(
      hs_bf, zwih_bf, z_b, zxg, 512, 1024);

  // z scan (means/logvars -> d_out; z bf16 -> dec_in cols 256..383)
  z_scan_kernel<<<1, 512, 32 * 1024, stream>>>(
      zxg, zwhh_bf, mw_bf, vw_bf, mean_b, logvar_b, eps, dec_in_bf,
      means, lvs, 64);

  // Decoder input projection: [2048,384] @ [2048,384]^T -> [2048,2048]
  gemm_bf16_kernel<<<dim3(16, 16), 256, 16384, stream>>>(
      dec_in_bf, dwih_bf, dec_b, dec_xg, 2048, 384);

  // Decoder scan -> dec_h_bf [2048][512] (grid=1; second param set unused)
  lstm_scan_kernel<<<1, 1024, 96 * 1024, stream>>>(
      dec_xg, dwhh_bf, dec_h_bf, 512, 0, 0,
      dec_xg, dwhh_bf, dec_h_bf, 512, 0, 0, 64);

  // Output projection: [2048,512] @ [32000,512]^T -> logits (dominant GEMM)
  gemm_bf16_kernel<<<dim3(250, 16), 256, 16384, stream>>>(
      dec_h_bf, outw_bf, out_b, logits, 32000, 512);
}
